// MultiHeadAttention_47201690583512
// MI455X (gfx1250) — matrix operand order repaired and checked by reading, not compiled, not run on hardware
//
#include <hip/hip_runtime.h>

// ---------------------------------------------------------------------------
// MultiHeadAttention for MI455X (gfx1250, wave32, WMMA + TDM).
//   B=4, S=1500 (padded 1504), C=1024, H=16, D=64.
// f32->f16 convert -> Q/K/V proj (WMMA GEMM, W tile TDM-staged in LDS,
// double buffered, 32 M-rows/wave) -> flash attention (K/V tiles TDM-staged,
// WMMA scores, DPP16 online softmax, WMMA PV) -> output proj.
// ---------------------------------------------------------------------------

typedef __attribute__((ext_vector_type(16))) _Float16     v16h;
typedef __attribute__((ext_vector_type(8)))  _Float16     v8h;
typedef __attribute__((ext_vector_type(8)))  float        v8f;
typedef __attribute__((ext_vector_type(4)))  unsigned int v4u;
typedef __attribute__((ext_vector_type(8)))  int          v8i;
typedef __attribute__((ext_vector_type(4)))  int          v4i;

#define BATCH   4
#define SEQ     1500
#define SEQP    1504
#define CDIM    1024
#define NHEAD   16
#define HDIM    64
#define MTOT    (BATCH * SEQ)       // 6000 rows (divisible by 16)
#define NMTILE  (MTOT / 16)         // 375 16-row tiles
#define NM32    ((NMTILE + 1) / 2)  // 188 32-row tiles
#define NQTILE  ((SEQ + 15) / 16)   // 94

#if defined(__has_builtin)
#if __has_builtin(__builtin_amdgcn_tensor_load_to_lds) && \
    __has_builtin(__builtin_amdgcn_s_wait_tensorcnt)
#define HAVE_TDM 1
#endif
#endif
#ifndef HAVE_TDM
#define HAVE_TDM 0
#endif

// ------------------------------- vector utils ------------------------------
__device__ __forceinline__ v16h make_v16(v8h lo, v8h hi) {
  v16h r;
#pragma unroll
  for (int i = 0; i < 8; ++i) { r[i] = lo[i]; r[i + 8] = hi[i]; }
  return r;
}
__device__ __forceinline__ v16h load_a16(const _Float16* p0, const _Float16* p1) {
  return make_v16(*(const v8h*)p0, *(const v8h*)p1);
}
__device__ __forceinline__ v16h load_b16(const _Float16* p) {
  return make_v16(*(const v8h*)p, *(const v8h*)(p + 8));
}
__device__ __forceinline__ v8f wmma_f16(v16h a, v16h b, v8f c) {
  return __builtin_amdgcn_wmma_f32_16x16x32_f16(false, a, false, b,
                                                (short)0, c, false, false);
}

// ------------------------- DPP16 row reductions ----------------------------
template <int CTRL>
__device__ __forceinline__ float dpp_mov(float x) {
  int i = __builtin_bit_cast(int, x);
  int y = __builtin_amdgcn_update_dpp(i, i, CTRL, 0xF, 0xF, false);
  return __builtin_bit_cast(float, y);
}
__device__ __forceinline__ float row_max16(float x) {
  x = fmaxf(x, dpp_mov<0xB1>(x));    // quad_perm xor1
  x = fmaxf(x, dpp_mov<0x4E>(x));    // quad_perm xor2
  x = fmaxf(x, dpp_mov<0x141>(x));   // row_half_mirror
  x = fmaxf(x, dpp_mov<0x140>(x));   // row_mirror
  return x;
}
__device__ __forceinline__ float row_sum16(float x) {
  x += dpp_mov<0xB1>(x);
  x += dpp_mov<0x4E>(x);
  x += dpp_mov<0x141>(x);
  x += dpp_mov<0x140>(x);
  return x;
}

// --------------------------- TDM tile staging ------------------------------
#if HAVE_TDM
// 2D tile -> LDS via Tensor Data Mover (D# per CDNA5 ISA).
// data_size=8B units; pad_enable inserts (pad_amt+1)*4B after each
// 2^(pad_int+1) DWORDs so ds_load_b128 column reads spread across banks.
// lds_off: absolute LDS byte offset (single __shared__ object => base 0).
__device__ __forceinline__ void tdm_load_2d(unsigned lds_off, const void* g,
                                            unsigned width_units, unsigned rows,
                                            unsigned stride_units,
                                            unsigned pad_int_code,
                                            unsigned pad_amt_code) {
  unsigned long long ga = (unsigned long long)(uintptr_t)g;
  v4u g0;
  g0[0] = 1u;                                         // count=1, user mode
  g0[1] = lds_off;                                    // lds_addr
  g0[2] = (unsigned)ga;                               // global_addr[31:0]
  g0[3] = (unsigned)((ga >> 32) & 0x01FFFFFFull) | (2u << 30);  // hi | type=2
  v8i g1;
  g1[0] = (int)((3u << 16) | (1u << 20) |             // data_size=8B, pad_enable
                (pad_int_code << 22) | (pad_amt_code << 25));
  g1[1] = (int)((width_units & 0xFFFFu) << 16);       // tensor_dim0 lo
  g1[2] = (int)((width_units >> 16) | ((rows & 0xFFFFu) << 16));
  g1[3] = (int)((rows >> 16) | ((width_units & 0xFFFFu) << 16));
  g1[4] = (int)(rows & 0xFFFFu);                      // tile_dim1 (tile_dim2=0)
  g1[5] = (int)stride_units;                          // tensor_dim0_stride lo
  g1[6] = 0;
  g1[7] = 0;
  v4i z4 = {0, 0, 0, 0};
#if __clang_major__ >= 23
  v8i z8 = (v8i)0;
  __builtin_amdgcn_tensor_load_to_lds(g0, g1, z4, z4, z8, 0);
#else
  __builtin_amdgcn_tensor_load_to_lds(g0, g1, z4, z4, 0);
#endif
}
#endif

__device__ __forceinline__ void stage_tile(unsigned lds_byte_off,
                                           _Float16* lds_ptr,
                                           const _Float16* src,
                                           int rows, int row_halves,
                                           int src_stride_halves,
                                           int dst_stride_halves,
                                           unsigned pad_int_code,
                                           unsigned pad_amt_code,
                                           int tid, int wave) {
#if HAVE_TDM
  if (wave == 0)
    tdm_load_2d(lds_byte_off, (const void*)src,
                (unsigned)(row_halves / 4), (unsigned)rows,
                (unsigned)(src_stride_halves / 4), pad_int_code, pad_amt_code);
  (void)lds_ptr; (void)dst_stride_halves; (void)tid;
#else
  const int cpr = row_halves / 8;
  const int total = rows * cpr;
  for (int c = tid; c < total; c += 128) {
    const int r = c / cpr, q = c - r * cpr;
    *(v8h*)(lds_ptr + r * dst_stride_halves + q * 8) =
        *(const v8h*)(src + r * src_stride_halves + q * 8);
  }
  (void)lds_byte_off; (void)pad_int_code; (void)pad_amt_code; (void)wave;
#endif
}

__device__ __forceinline__ void tensor_wait0(int wave) {
#if HAVE_TDM
  if (wave == 0) __builtin_amdgcn_s_wait_tensorcnt(0);
#endif
  (void)wave;
}

// ------------------------------ f32 -> f16 ---------------------------------
__global__ void cvt_f32_to_f16(const float* __restrict__ src,
                               _Float16* __restrict__ dst, int n) {
  int i = blockIdx.x * blockDim.x + threadIdx.x;
  if (i < n) dst[i] = (_Float16)src[i];
}

// ---------------------------------------------------------------------------
// GEMM: C[6000,1024] = A * W^T (+bias). Block = 4 waves sharing a 64x32 W
// tile (TDM, double buffered, padded rows). Each wave: 32(M) x 64(N),
// two A streams prefetched one K-step ahead, B tiles loaded as a batch.
// ---------------------------------------------------------------------------
#define WROW 40                 // padded LDS row stride in halves (64B + 16B)
#define WBUF (64 * WROW)        // halves per buffer (5120 B)

template <int MODE>
__global__ __launch_bounds__(128, 1)
void gemm_wmma_f16(const _Float16* __restrict__ A,
                   const _Float16* __restrict__ W,
                   const float* __restrict__ bias,
                   void* __restrict__ dst) {
  __shared__ alignas(16) _Float16 wsm[2 * WBUF];
  const int lane = threadIdx.x;
  const int sel  = lane >> 4;
  const int lm   = lane & 15;
  const int w    = threadIdx.y;
  const int tid  = w * 32 + lane;
  const int n0   = blockIdx.x * 64;
  int mt = blockIdx.y * 4 + w;
  if (mt >= NM32) mt = NM32 - 1;
  int m0 = mt * 32;
  if (m0 > MTOT - 32) m0 = MTOT - 32;   // overlap rows rewrite identical data

  const _Float16* arow0 = A + (m0 + lm) * CDIM;
  const _Float16* arow1 = A + (m0 + 16 + lm) * CDIM;

  stage_tile(0, wsm, W + n0 * CDIM, 64, 32, CDIM, WROW, 3, 3, tid, w);

  v8f acc[2][4] = {};
  v16h a0n = load_a16(arow0 + sel * 16, arow0 + 8 + sel * 16);
  v16h a1n = load_a16(arow1 + sel * 16, arow1 + 8 + sel * 16);

  constexpr int NK = CDIM / 32;
  for (int it = 0; it < NK; ++it) {
    const int kk = it * 32;
    _Float16* buf = wsm + (it & 1) * WBUF;
    tensor_wait0(w);
    __syncthreads();
    if (it + 1 < NK) {
      const int nb = (it + 1) & 1;
      stage_tile((unsigned)(nb * WBUF * 2), wsm + nb * WBUF,
                 W + n0 * CDIM + kk + 32, 64, 32, CDIM, WROW, 3, 3, tid, w);
    }
    const v16h a0 = a0n, a1 = a1n;
    if (it + 1 < NK) {
      a0n = load_a16(arow0 + kk + 32 + sel * 16, arow0 + kk + 40 + sel * 16);
      a1n = load_a16(arow1 + kk + 32 + sel * 16, arow1 + kk + 40 + sel * 16);
    }
    // batch the B-tile LDS reads so DScnt waits can be partial
    v16h bt[4];
#pragma unroll
    for (int t = 0; t < 4; ++t)
      bt[t] = load_b16(buf + (t * 16 + lm) * WROW + sel * 16);
#pragma unroll
    for (int t = 0; t < 4; ++t) acc[0][t] = wmma_f16(a0, bt[t], acc[0][t]);
#pragma unroll
    for (int t = 0; t < 4; ++t) acc[1][t] = wmma_f16(a1, bt[t], acc[1][t]);
  }

#pragma unroll
  for (int half = 0; half < 2; ++half) {
#pragma unroll
    for (int t = 0; t < 4; ++t) {
      const int n  = n0 + t * 16 + lm;
      const float bv = bias ? bias[n] : 0.0f;
#pragma unroll
      for (int r = 0; r < 8; ++r) {
        const int m = m0 + half * 16 + r + sel * 8;
        const float v = acc[half][t][r] + bv;
        if (MODE == 0) {
          ((float*)dst)[m * CDIM + n] = v;
        } else {
          const int b = m / SEQ, s = m - b * SEQ;
          const int h = n >> 6, d = n & 63;
          const int bh = b * NHEAD + h;
          if (MODE == 1)
            ((_Float16*)dst)[(bh * SEQP + s) * HDIM + d] = (_Float16)v;
          else
            ((_Float16*)dst)[(bh * HDIM + d) * SEQP + s] = (_Float16)v;
        }
      }
    }
  }
}

// ---------------------------------------------------------------------------
// Flash attention: block = 4 waves sharing (b,h); each wave owns a 16-q tile.
// K/V tiles TDM-staged (double buffered) shared by the 4 waves. V-tile LDS
// reads issued right after the P stores: short live range (no spills), still
// ahead of the pa reload in the in-order DS queue.
// ---------------------------------------------------------------------------
#define KROW 72                      // 128B row + 16B pad (halves)
#define KBUF (32 * KROW * 2)         // bytes per K buffer (4608)
#define VROW 40                      // 64B row + 16B pad (halves)
#define VBUF (64 * VROW * 2)         // bytes per V buffer (5120)
#define K_OFF 0
#define V_OFF (2 * KBUF)             // 9216
#define P_OFF (V_OFF + 2 * VBUF)     // 19456
#define SMEM_TOTAL (P_OFF + 4 * 1024)

__global__ __launch_bounds__(128, 1)
void attn_flash_wmma(const _Float16* __restrict__ Qh,
                     const _Float16* __restrict__ Kh,
                     const _Float16* __restrict__ Vt,
                     _Float16* __restrict__ Oh) {
  __shared__ alignas(16) char smem[SMEM_TOTAL];
  const int lane = threadIdx.x;
  const int sel  = lane >> 4;
  const int lm   = lane & 15;
  const int w    = threadIdx.y;
  const int tid  = w * 32 + lane;
  const int bh   = blockIdx.x;
  int qt = blockIdx.y * 4 + w;
  if (qt >= NQTILE) qt = NQTILE - 1;   // tail waves duplicate (identical writes)
  const int q0 = qt * 16;

  const _Float16* qbase = Qh + bh * SEQP * HDIM;
  const _Float16* kbase = Kh + bh * SEQP * HDIM;
  const _Float16* vbase = Vt + bh * HDIM * SEQP;
  _Float16* pl = (_Float16*)(smem + P_OFF + w * 1024);

  int qrow = q0 + lm; if (qrow > SEQ - 1) qrow = SEQ - 1;
  const _Float16* qp = qbase + qrow * HDIM;
  const v16h qa0 = load_a16(qp + sel * 16,      qp + 8 + sel * 16);
  const v16h qa1 = load_a16(qp + 32 + sel * 16, qp + 40 + sel * 16);

  v8f acc[4] = {};
  float mrow[8], lrow[8];
#pragma unroll
  for (int r = 0; r < 8; ++r) { mrow[r] = -3.0e30f; lrow[r] = 0.0f; }

  stage_tile(K_OFF, (_Float16*)(smem + K_OFF), kbase, 32, 64, HDIM, KROW, 4, 3, tid, w);
  stage_tile(V_OFF, (_Float16*)(smem + V_OFF), vbase, 64, 32, SEQP, VROW, 3, 3, tid, w);

  constexpr int NJ = SEQP / 32;   // 47
  for (int it = 0; it < NJ; ++it) {
    const int jt  = it * 32;
    const int cur = it & 1;
    _Float16* klds = (_Float16*)(smem + K_OFF + cur * KBUF);
    _Float16* vlds = (_Float16*)(smem + V_OFF + cur * VBUF);
    tensor_wait0(w);
    __syncthreads();
    if (it + 1 < NJ) {
      const int nb = cur ^ 1;
      stage_tile((unsigned)(K_OFF + nb * KBUF), (_Float16*)(smem + K_OFF + nb * KBUF),
                 kbase + (jt + 32) * HDIM, 32, 64, HDIM, KROW, 4, 3, tid, w);
      stage_tile((unsigned)(V_OFF + nb * VBUF), (_Float16*)(smem + V_OFF + nb * VBUF),
                 vbase + (jt + 32), 64, 32, SEQP, VROW, 3, 3, tid, w);
    }

    // all K-tile LDS reads first, then the score WMMA chain
    v16h kb[4];
    kb[0] = load_b16(klds + lm * KROW + sel * 16);
    kb[1] = load_b16(klds + lm * KROW + 32 + sel * 16);
    kb[2] = load_b16(klds + (16 + lm) * KROW + sel * 16);
    kb[3] = load_b16(klds + (16 + lm) * KROW + 32 + sel * 16);
    v8f s01[2] = {};
    s01[0] = wmma_f16(qa0, kb[0], s01[0]);
    s01[0] = wmma_f16(qa1, kb[1], s01[0]);
    s01[1] = wmma_f16(qa0, kb[2], s01[1]);
    s01[1] = wmma_f16(qa1, kb[3], s01[1]);

    // scale + key-padding mask
#pragma unroll
    for (int u = 0; u < 2; ++u) {
      const float maskadd = (jt + u * 16 + lm < SEQ) ? 0.0f : -3.0e30f;
#pragma unroll
      for (int r = 0; r < 8; ++r) s01[u][r] = s01[u][r] * 0.125f + maskadd;
    }

    // online softmax (row r lives in the 16-lane half `sel`)
#pragma unroll
    for (int r = 0; r < 8; ++r) {
      const float tmax = row_max16(fmaxf(s01[0][r], s01[1][r]));
      const float mn = fmaxf(mrow[r], tmax);
      const float alpha = __expf(mrow[r] - mn);
      mrow[r] = mn;
      const float p0 = __expf(s01[0][r] - mn);
      const float p1 = __expf(s01[1][r] - mn);
      lrow[r] = lrow[r] * alpha + row_sum16(p0 + p1);
#pragma unroll
      for (int t4 = 0; t4 < 4; ++t4) acc[t4][r] *= alpha;
      const int row = r + sel * 8;
      pl[row * 32 + lm]      = (_Float16)p0;
      pl[row * 32 + 16 + lm] = (_Float16)p1;
    }

    // V-tile reads: short live range, queued ahead of the pa reload
    v16h vb[4];
#pragma unroll
    for (int t4 = 0; t4 < 4; ++t4)
      vb[t4] = load_b16(vlds + (t4 * 16 + lm) * VROW + sel * 16);

    // per-wave LDS is in-order; just stop compiler reordering.
    asm volatile("" ::: "memory");
    const v16h pa = load_a16(pl + lm * 32 + sel * 16, pl + lm * 32 + 8 + sel * 16);
    asm volatile("" ::: "memory");

#pragma unroll
    for (int t4 = 0; t4 < 4; ++t4) acc[t4] = wmma_f16(pa, vb[t4], acc[t4]);
  }

  const int b = bh >> 4, h = bh & 15;
#pragma unroll
  for (int r = 0; r < 8; ++r) {
    const int m = q0 + r + sel * 8;
    if (m < SEQ) {
      const float inv = 1.0f / lrow[r];
#pragma unroll
      for (int t4 = 0; t4 < 4; ++t4)
        Oh[(b * SEQ + m) * CDIM + h * HDIM + t4 * 16 + lm] =
            (_Float16)(acc[t4][r] * inv);
    }
  }
}

// ---------------------------------------------------------------------------
// Launch
// ---------------------------------------------------------------------------
extern "C" void kernel_launch(void* const* d_in, const int* in_sizes, int n_in,
                              void* d_out, int out_size, void* d_ws, size_t ws_size,
                              hipStream_t stream) {
  const float* x  = (const float*)d_in[0];
  const float* Wq = (const float*)d_in[1];
  const float* bq = (const float*)d_in[2];
  const float* Wk = (const float*)d_in[3];
  const float* Wv = (const float*)d_in[4];
  const float* bv = (const float*)d_in[5];
  const float* Wo = (const float*)d_in[6];
  const float* bo = (const float*)d_in[7];
  float* out = (float*)d_out;

  _Float16* ws = (_Float16*)d_ws;
  size_t off = 0;
  _Float16* xh  = ws + off; off += (size_t)MTOT * CDIM;
  _Float16* Wqh = ws + off; off += (size_t)CDIM * CDIM;
  _Float16* Wkh = ws + off; off += (size_t)CDIM * CDIM;
  _Float16* Wvh = ws + off; off += (size_t)CDIM * CDIM;
  _Float16* Woh = ws + off; off += (size_t)CDIM * CDIM;
  _Float16* Qh  = ws + off; off += (size_t)BATCH * NHEAD * SEQP * HDIM;
  _Float16* Khb = ws + off; off += (size_t)BATCH * NHEAD * SEQP * HDIM;
  _Float16* Vt  = ws + off; off += (size_t)BATCH * NHEAD * HDIM * SEQP;
  _Float16* Oh  = ws + off; off += (size_t)MTOT * CDIM;

  {
    const int nx = MTOT * CDIM, nw = CDIM * CDIM, tb = 256;
    cvt_f32_to_f16<<<(nx + tb - 1) / tb, tb, 0, stream>>>(x,  xh,  nx);
    cvt_f32_to_f16<<<(nw + tb - 1) / tb, tb, 0, stream>>>(Wq, Wqh, nw);
    cvt_f32_to_f16<<<(nw + tb - 1) / tb, tb, 0, stream>>>(Wk, Wkh, nw);
    cvt_f32_to_f16<<<(nw + tb - 1) / tb, tb, 0, stream>>>(Wv, Wvh, nw);
    cvt_f32_to_f16<<<(nw + tb - 1) / tb, tb, 0, stream>>>(Wo, Woh, nw);
  }

  const dim3 gblk(32, 4);
  const dim3 ggrd(CDIM / 64, (NM32 + 3) / 4);   // 16 x 47
  gemm_wmma_f16<1><<<ggrd, gblk, 0, stream>>>(xh, Wqh, bq, Qh);
  gemm_wmma_f16<1><<<ggrd, gblk, 0, stream>>>(xh, Wkh, (const float*)nullptr, Khb);
  gemm_wmma_f16<2><<<ggrd, gblk, 0, stream>>>(xh, Wvh, bv, Vt);

  attn_flash_wmma<<<dim3(BATCH * NHEAD, (NQTILE + 3) / 4), gblk, 0, stream>>>(
      Qh, Khb, Vt, Oh);

  gemm_wmma_f16<0><<<ggrd, gblk, 0, stream>>>(Oh, Woh, bo, out);
}
